// YOROLayer_55422257987676
// MI455X (gfx1250) — matrix operand order repaired and checked by reading, not compiled, not run on hardware
//
#include <hip/hip_runtime.h>
#include <hip/hip_bf16.h>
#include <math.h>

// ---------------------------------------------------------------------------
// Problem constants (match reference)
// ---------------------------------------------------------------------------
#define BATCH       16
#define IN_CH       256
#define FMAP        80          // 80x80
#define PADW        82          // padded 82x82
#define NUM_CLASSES 20
#define DEG_DEPTH   37
#define GROUP_DEPTH 99          // 1 + 20 + 4 + 2*37
#define N_ANCHOR    4
#define FMAP_DEPTH  396         // 99*4
#define CH_PAD      416         // padded to 26 tiles of 16 (13 macro-tiles of 32)
#define N_CT16      26
#define N_CT32      13
#define PIX_PER_B   6400        // 80*80
#define N_PIX       (BATCH * PIX_PER_B)          // 102400
#define TAP_ELEMS   4096        // one 16ch x 256K tap block (bf16 elems) = 8 KB

typedef __attribute__((ext_vector_type(16))) __bf16 v16bf;
typedef __attribute__((ext_vector_type(8)))  float  v8f;
typedef __attribute__((ext_vector_type(4)))  float  v4f;

// ---------------------------------------------------------------------------
// Stage 1: NCHW fp32 -> padded NHWC bf16  [B][82][82][256]
// ---------------------------------------------------------------------------
__global__ void pad_nhwc_bf16(const float* __restrict__ in, __bf16* __restrict__ out) {
    size_t t = (size_t)blockIdx.x * blockDim.x + threadIdx.x;
    int c = (int)(t & 255);
    size_t s = t >> 8;
    int x = (int)(s % PADW); s /= PADW;
    int y = (int)(s % PADW); s /= PADW;
    int b = (int)s;
    float v = 0.0f;
    if (x >= 1 && x <= FMAP && y >= 1 && y <= FMAP) {
        v = in[(((size_t)b * IN_CH + c) * FMAP + (y - 1)) * FMAP + (x - 1)];
    }
    out[t] = (__bf16)v;
}

// ---------------------------------------------------------------------------
// Stage 2: weight repack fp32 OIHW -> bf16 A-matrix lane layout
// layout: [cTile16(26)][tap(9)][kc(8)][lane(32)][j(16)]
// 16-bit A-matrix 16x32 layout (contiguous K order, matches B fill):
//   lanes 0-15 : M=lane,    K = j
//   lanes 16-31: M=lane-16, K = j+16
// ---------------------------------------------------------------------------
__global__ void repack_weights(const float* __restrict__ W, __bf16* __restrict__ wPack) {
    int t = blockIdx.x * blockDim.x + threadIdx.x;      // 0 .. 958463
    int idx = t;
    int j    = idx & 15;  idx >>= 4;
    int lane = idx & 31;  idx >>= 5;
    int kc   = idx & 7;   idx >>= 3;
    int tap  = idx % 9;   idx /= 9;
    int cTile = idx;                                    // 0..25
    int K = (lane < 16) ? j
                        : j + 16;
    int ci = kc * 32 + K;
    int co = cTile * 16 + (lane & 15);
    int ky = tap / 3, kx = tap % 3;
    float v = 0.0f;
    if (co < FMAP_DEPTH) {
        v = W[(((size_t)co * IN_CH + ci) * 3 + ky) * 3 + kx];
    }
    wPack[t] = (__bf16)v;
}

// ---------------------------------------------------------------------------
// Async global -> LDS staging (ASYNCcnt-tracked, gfx1250)
// ---------------------------------------------------------------------------
__device__ __forceinline__ void async_ld_b128(unsigned ldsAddr, unsigned voff,
                                              const void* sbase) {
    asm volatile("global_load_async_to_lds_b128 %0, %1, %2"
                 :: "v"(ldsAddr), "v"(voff), "s"(sbase)
                 : "memory");
}
__device__ __forceinline__ void wait_asynccnt_4() {
    asm volatile("s_wait_asynccnt 0x4" ::: "memory");
}
__device__ __forceinline__ void wait_asynccnt_0() {
    asm volatile("s_wait_asynccnt 0x0" ::: "memory");
}

// ---------------------------------------------------------------------------
// Stage 3: implicit-GEMM conv with WMMA bf16 -> f32
// grid:  x = 400 pixel groups (256 px), y = 13 macro channel tiles (32 ch)
// block: 256 threads = 8 waves; each wave: 32 out-ch x 32 px (2x2 accumulators)
// Weights double-buffered in LDS via global_load_async_to_lds_b128.
// convOut layout: [b][pixel][chPad=416] f32 (channels contiguous for decode)
// ---------------------------------------------------------------------------
__global__ void __launch_bounds__(256)
conv_wmma(const __bf16* __restrict__ inPad,
          const __bf16* __restrict__ wPack,
          float* __restrict__ convOut) {
    // two buffers x (two 16-ch tap blocks) = 2 x 16 KB
    __shared__ __attribute__((aligned(16))) __bf16 sW[2][2 * TAP_ELEMS];

    const int tid  = threadIdx.x;
    const int wave = tid >> 5;
    const int lane = tid & 31;
    const int ct0  = blockIdx.y * 2;          // first 16-ch tile of the pair

    // This wave's 32 pixels (two 16-px subtiles). 6400 % 256 == 0 -> one batch.
    const int gpx  = blockIdx.x * 256 + wave * 32;
    const int bIdx = gpx / PIX_PER_B;
    const int pin  = gpx % PIX_PER_B;
    const int lp   = lane & 15;           // pixel (N) within subtile
    const int hi   = lane >> 4;           // K-half / M-half selector

    const int px0 = pin + lp;
    const int px1 = pin + 16 + lp;
    const int y0 = px0 / FMAP, x0 = px0 % FMAP;
    const int y1 = px1 / FMAP, x1 = px1 % FMAP;

    const size_t bOff = (size_t)bIdx * PADW * PADW * IN_CH;
    const __bf16* p0 = inPad + bOff + ((size_t)y0 * PADW + x0) * IN_CH + hi * 16;
    const __bf16* p1 = inPad + bOff + ((size_t)y1 * PADW + x1) * IN_CH + hi * 16;

    const __bf16* w0 = wPack + (size_t)(ct0 + 0) * (9 * TAP_ELEMS);
    const __bf16* w1 = wPack + (size_t)(ct0 + 1) * (9 * TAP_ELEMS);

    const unsigned ldsB0 = (unsigned)(uintptr_t)(void*)&sW[0][0];
    const unsigned ldsB1 = (unsigned)(uintptr_t)(void*)&sW[1][0];
    const unsigned vo0 = (unsigned)tid * 16u;          // byte offset, chunk tid
    const unsigned vo1 = vo0 + 4096u;                  // byte offset, chunk tid+256

    // stage tap-block of both 16-ch tiles into buffer `buf` (4 async b128/thread)
    auto stage = [&](int buf, int tap) {
        const unsigned lb = buf ? ldsB1 : ldsB0;
        const void* s0 = (const void*)(w0 + (size_t)tap * TAP_ELEMS);
        const void* s1 = (const void*)(w1 + (size_t)tap * TAP_ELEMS);
        async_ld_b128(lb + vo0,        vo0, s0);
        async_ld_b128(lb + vo1,        vo1, s0);
        async_ld_b128(lb + 8192u + vo0, vo0, s1);
        async_ld_b128(lb + 8192u + vo1, vo1, s1);
    };

    v8f acc00 = {}, acc01 = {}, acc10 = {}, acc11 = {};

    stage(0, 0);                                   // prologue
    for (int tap = 0; tap < 9; ++tap) {
        if (tap < 8) {                             // prefetch next tap's weights
            stage((tap + 1) & 1, tap + 1);
            wait_asynccnt_4();                     // this tap's 4 ops are done
        } else {
            wait_asynccnt_0();
        }
        __syncthreads();                           // all waves' stages visible

        const __bf16* sw = &sW[tap & 1][0];
        const int ky = tap / 3, kx = tap % 3;
        const size_t tOff = ((size_t)ky * PADW + kx) * IN_CH;

#pragma unroll
        for (int kc = 0; kc < 8; ++kc) {
            v16bf a0 = *reinterpret_cast<const v16bf*>(&sw[(kc * 32 + lane) * 16]);
            v16bf a1 = *reinterpret_cast<const v16bf*>(&sw[TAP_ELEMS + (kc * 32 + lane) * 16]);
            v16bf f0 = *reinterpret_cast<const v16bf*>(p0 + tOff + kc * 32);
            v16bf f1 = *reinterpret_cast<const v16bf*>(p1 + tOff + kc * 32);
            acc00 = __builtin_amdgcn_wmma_f32_16x16x32_bf16(
                false, a0, false, f0, (short)0, acc00, false, false);
            acc01 = __builtin_amdgcn_wmma_f32_16x16x32_bf16(
                false, a0, false, f1, (short)0, acc01, false, false);
            acc10 = __builtin_amdgcn_wmma_f32_16x16x32_bf16(
                false, a1, false, f0, (short)0, acc10, false, false);
            acc11 = __builtin_amdgcn_wmma_f32_16x16x32_bf16(
                false, a1, false, f1, (short)0, acc11, false, false);
        }
        __syncthreads();                           // reads done before refill
    }

    // Store: C/D layout -> lane holds N=lane&15, rows M = vgpr + 8*hi.
    const int chA = (ct0 + 0) * 16 + hi * 8;
    const int chB = (ct0 + 1) * 16 + hi * 8;
    float* r0 = convOut + ((size_t)(bIdx * PIX_PER_B + px0)) * CH_PAD;
    float* r1 = convOut + ((size_t)(bIdx * PIX_PER_B + px1)) * CH_PAD;
    {
        const v4f* h;
        h = reinterpret_cast<const v4f*>(&acc00);
        reinterpret_cast<v4f*>(r0 + chA)[0] = h[0];
        reinterpret_cast<v4f*>(r0 + chA)[1] = h[1];
        h = reinterpret_cast<const v4f*>(&acc01);
        reinterpret_cast<v4f*>(r1 + chA)[0] = h[0];
        reinterpret_cast<v4f*>(r1 + chA)[1] = h[1];
        h = reinterpret_cast<const v4f*>(&acc10);
        reinterpret_cast<v4f*>(r0 + chB)[0] = h[0];
        reinterpret_cast<v4f*>(r0 + chB)[1] = h[1];
        h = reinterpret_cast<const v4f*>(&acc11);
        reinterpret_cast<v4f*>(r1 + chB)[0] = h[0];
        reinterpret_cast<v4f*>(r1 + chB)[1] = h[1];
    }
}

// ---------------------------------------------------------------------------
// Stage 4: decode. one thread per (b, anchor, pixel) = 409,600 threads.
// ---------------------------------------------------------------------------
__device__ __forceinline__ float sigmoidf_(float z) {
    return 1.0f / (1.0f + __expf(-z));
}

__global__ void decode_kernel(const float* __restrict__ conv,
                              const float* __restrict__ bias,
                              float* __restrict__ out) {
    int t = blockIdx.x * blockDim.x + threadIdx.x;
    if (t >= BATCH * N_ANCHOR * PIX_PER_B) return;

    const int bIdx = t / (N_ANCHOR * PIX_PER_B);
    const int r    = t % (N_ANCHOR * PIX_PER_B);     // anchor*6400 + pixel
    const int a    = r / PIX_PER_B;
    const int px   = r % PIX_PER_B;
    const int yy   = px / FMAP;
    const int xx   = px % FMAP;

    const float* v  = conv + ((size_t)(bIdx * PIX_PER_B + px)) * CH_PAD + a * GROUP_DEPTH;
    const float* bb = bias + a * GROUP_DEPTH;

    float conf = sigmoidf_(v[0] + bb[0]);

    int clsIdx = 0;
    float clsMax = v[1] + bb[1];
    for (int i = 1; i < NUM_CLASSES; ++i) {
        float c = v[1 + i] + bb[1 + i];
        if (c > clsMax) { clsMax = c; clsIdx = i; }
    }

    float bx = sigmoidf_(v[21] + bb[21]);
    float by = sigmoidf_(v[22] + bb[22]);
    float bw = v[23] + bb[23];
    float bh = v[24] + bb[24];

    int degIdx = 0;
    float degMax = v[25] + bb[25];
    for (int i = 1; i < DEG_DEPTH; ++i) {
        float c = v[25 + i] + bb[25 + i];
        if (c > degMax) { degMax = c; degIdx = i; }
    }
    float shift = v[62 + degIdx] + bb[62 + degIdx];
    float predDeg = (-180.0f + 10.0f * degIdx) + shift * 5.0f;

    const float anchorPix[4] = {16.0f, 32.0f, 64.0f, 128.0f};
    float boxX = (bx + (float)xx) * 8.0f;
    float boxY = (by + (float)yy) * 8.0f;
    float boxW = __expf(bw) * anchorPix[a];
    float boxH = __expf(bh) * anchorPix[a];

    const int NP = N_ANCHOR * PIX_PER_B;             // 25600
    float* outConf  = out;
    float* outClass = out + (size_t)BATCH * NP;
    float* outBoxes = out + (size_t)2 * BATCH * NP;
    float* outDeg   = out + (size_t)2 * BATCH * NP + (size_t)4 * BATCH * NP;

    const size_t oi = (size_t)bIdx * NP + r;
    outConf[oi]  = conf;
    outClass[oi] = (float)clsIdx;
    outBoxes[oi * 4 + 0] = boxX;
    outBoxes[oi * 4 + 1] = boxY;
    outBoxes[oi * 4 + 2] = boxW;
    outBoxes[oi * 4 + 3] = boxH;
    outDeg[oi]   = predDeg;
}

// ---------------------------------------------------------------------------
// Launch
// ---------------------------------------------------------------------------
extern "C" void kernel_launch(void* const* d_in, const int* in_sizes, int n_in,
                              void* d_out, int out_size, void* d_ws, size_t ws_size,
                              hipStream_t stream) {
    const float* inputs = (const float*)d_in[0];   // [16,256,80,80] f32
    const float* W      = (const float*)d_in[1];   // [396,256,3,3]  f32
    const float* bias   = (const float*)d_in[2];   // [396]          f32
    float* out = (float*)d_out;

    char* ws = (char*)d_ws;
    const size_t padBytes   = (size_t)BATCH * PADW * PADW * IN_CH * 2;      // 55,083,008
    const size_t wPackBytes = (size_t)N_CT16 * 9 * 8 * 32 * 16 * 2;         //  1,916,928
    __bf16* inPad   = (__bf16*)ws;
    __bf16* wPack   = (__bf16*)(ws + padBytes);
    float*  convOut = (float*)(ws + padBytes + wPackBytes);                 // 170.4 MB

    // Stage 1: pad/transpose/convert
    {
        const size_t n = (size_t)BATCH * PADW * PADW * IN_CH;               // 27,541,504
        pad_nhwc_bf16<<<dim3((unsigned)(n / 256)), dim3(256), 0, stream>>>(inputs, inPad);
    }
    // Stage 2: weight repack
    {
        const int n = N_CT16 * 9 * 8 * 32 * 16;                             // 958,464
        repack_weights<<<dim3(n / 256), dim3(256), 0, stream>>>(W, wPack);
    }
    // Stage 3: WMMA conv
    {
        conv_wmma<<<dim3(N_PIX / 256, N_CT32), dim3(256), 0, stream>>>(inPad, wPack, convOut);
    }
    // Stage 4: decode
    {
        const int n = BATCH * N_ANCHOR * PIX_PER_B;                         // 409,600
        decode_kernel<<<dim3((n + 255) / 256), dim3(256), 0, stream>>>(convOut, bias, out);
    }
}
